// BahdanauAttention_89154931130659
// MI455X (gfx1250) — matrix operand order repaired and checked by reading, not compiled, not run on hardware
//
#include <hip/hip_runtime.h>
#include <hip/hip_bf16.h>
#include <math.h>

typedef __attribute__((ext_vector_type(16))) __bf16 v16bf;
typedef __attribute__((ext_vector_type(8)))  float  v8f;
typedef __attribute__((ext_vector_type(4)))  unsigned int u32x4;
typedef __attribute__((ext_vector_type(8)))  int i32x8;
typedef __attribute__((ext_vector_type(4)))  int i32x4;

#define Bsz 32
#define Tsz 2048
#define Hsz 1024
#define NEGVAL (-1e9f)

// ---------------------------------------------------------------- utilities
__global__ void zero_f32(float* __restrict__ p, int n) {
    int i = blockIdx.x * blockDim.x + threadIdx.x;
    if (i < n) p[i] = 0.0f;
}

// ---------------------------------------------------------------- W_enc pack
// Pack W_enc (f32, [o,h] row-major) into bf16 WMMA B-fragments.
// Fragment (nc, ks) covers outputs o = nc*16..+15, k = ks*32..+31, laid out as
// 1024 contiguous bytes (32 lanes x 32B).  Fragment address = (nc*32+ks)*1024.
__global__ void pack_wenc(const float* __restrict__ W, __bf16* __restrict__ P) {
    int gid  = blockIdx.x * blockDim.x + threadIdx.x;   // 65536 threads
    int lane = gid & 31;
    int frag = gid >> 5;                                // 0..2047
    int nc   = frag >> 5;                               // 0..63
    int ks   = frag & 31;                               // 0..31
    int n    = lane & 15;
    int half = lane >> 4;
    __bf16* dst       = P + (size_t)frag * 512 + (size_t)lane * 16;
    const float* src  = W + (size_t)(nc * 16 + n) * Hsz + ks * 32 + half * 8;
#pragma unroll
    for (int e = 0; e < 16; ++e) dst[e] = (__bf16)src[e < 8 ? e : e + 8];
}

// ---------------------------------------------------------------- dec_p
__global__ void dec_proj(const float* __restrict__ dec, const float* __restrict__ Wd,
                         float* __restrict__ decp) {
    int o   = blockIdx.x;     // 0..1023
    int tid = threadIdx.x;    // 0..255
    float acc[Bsz];
#pragma unroll
    for (int b = 0; b < Bsz; ++b) acc[b] = 0.0f;
    for (int h = tid; h < Hsz; h += 256) {
        float wv = Wd[(size_t)o * Hsz + h];
#pragma unroll
        for (int b = 0; b < Bsz; ++b) acc[b] += wv * dec[(size_t)b * Hsz + h];
    }
    __shared__ float sd[256];
    for (int b = 0; b < Bsz; ++b) {
        sd[tid] = acc[b];
        __syncthreads();
        for (int st = 128; st > 0; st >>= 1) {
            if (tid < st) sd[tid] += sd[tid + st];
            __syncthreads();
        }
        if (tid == 0) decp[(size_t)b * Hsz + o] = sd[0];
        __syncthreads();
    }
}

// ---------------------------------------------------------------- TDM helper
// DMA one k-slab of packed W_enc into LDS:
//   2D tile: 32 rows (nc) x 1024 bytes (one fragment), row stride 32768 bytes.
// D# built per CDNA5 ISA 8.3/8.4: group0 = {count/flags, lds_addr, gaddr, type=2},
// group1 = {mask=0/data_size=1B, tensor dims, tile dims, strides}.
__device__ __forceinline__ void tdm_load_slab(const void* gsrc, unsigned lds_addr) {
    unsigned long long ga = (unsigned long long)(uintptr_t)gsrc;
    u32x4 g0;
    g0[0] = 1u;                                            // count=1, user mode
    g0[1] = lds_addr;                                      // LDS byte address
    g0[2] = (unsigned)(ga & 0xFFFFFFFFu);                  // global_addr[31:0]
    g0[3] = (unsigned)((ga >> 32) & 0x01FFFFFFu) | (2u << 30); // addr[56:32], type=2

    const unsigned td0 = 32768u;   // tensor_dim0 (1B units)
    const unsigned td1 = 64u;      // tensor_dim1
    const unsigned tile0 = 1024u;  // tile_dim0 = one fragment (bytes)
    const unsigned tile1 = 32u;    // tile_dim1 = 32 fragments (rows)
    const unsigned long long s0 = 32768ull;  // tensor_dim0_stride (bytes)
    i32x8 g1;
    g1[0] = 0;                                             // wg_mask=0, data_size=1B
    g1[1] = (int)((td0 & 0xFFFFu) << 16);                  // td0[15:0] @ bits 63:48
    g1[2] = (int)((td0 >> 16) | ((td1 & 0xFFFFu) << 16));  // td0[31:16], td1[15:0]
    g1[3] = (int)((td1 >> 16) | (tile0 << 16));            // td1[31:16], tile_dim0
    g1[4] = (int)(tile1);                                  // tile_dim1, tile_dim2=0
    g1[5] = (int)(s0 & 0xFFFFFFFFu);                       // stride0[31:0]
    g1[6] = (int)((s0 >> 32) & 0xFFFFu);                   // stride0[47:32]
    g1[7] = 0;
    i32x4 z4 = {0, 0, 0, 0};
    i32x8 z8 = {0, 0, 0, 0, 0, 0, 0, 0};
    __builtin_amdgcn_tensor_load_to_lds(g0, g1, z4, z4, z8, 0);
}

// ---------------------------------------------------------------- main WMMA
// Block = 32 t-rows x 512 outputs.  8 waves; wave w owns outputs
// hb*512 + w*64 .. +63 (4 N-subtiles) for 2 M-tiles -> 8 WMMAs per k-step.
// B slabs are TDM-DMA'd into LDS (double buffered), A converted f32->bf16 on load.
__global__ __launch_bounds__(256)
void attn_scores_wmma(const float* __restrict__ enc, const __bf16* __restrict__ Pw,
                      const float* __restrict__ decp, const float* __restrict__ v,
                      float* __restrict__ scores) {
    __shared__ __align__(16) char smem[2][32768];   // two 32 KB B-slab buffers

    int tile  = blockIdx.x;            // 0..2047 : (b, 32-row tile)
    int hb    = blockIdx.y;            // 0..1    : which half of H
    int b     = tile >> 6;             // 64 row-tiles per batch
    int trow  = (tile & 63) << 5;      // 32 rows per block
    int tid   = threadIdx.x;
    int lane  = tid & 31;
    int wave  = tid >> 5;              // 0..7
    int half  = lane >> 4;
    int nlane = lane & 15;

    // A rows for the two M-tiles (16-bit A layout: both halves cover same rows).
    const float* arow0 = enc + ((size_t)b * Tsz + trow + nlane) * Hsz + half * 8;
    const float* arow1 = arow0 + (size_t)16 * Hsz;

    v8f c[2][4] = {};
    // global fragment base for this block's h-half: nc_g = hb*32 + nc_local
    const __bf16* slab_base = Pw + (size_t)hb * 32 * 32 * 512;   // fragments (hb*32 .. +31, *)

    // prologue: DMA slab ks=0 into buffer 0
    if (wave == 0) tdm_load_slab(slab_base, (unsigned)(uintptr_t)&smem[0][0]);

    for (int ks = 0; ks < 32; ++ks) {
        int cur = ks & 1;
        if (wave == 0) __builtin_amdgcn_s_wait_tensorcnt(0);
        __syncthreads();                       // slab ks visible; prior reads done
        if (ks < 31 && wave == 0)
            tdm_load_slab(slab_base + (size_t)(ks + 1) * 512,
                          (unsigned)(uintptr_t)&smem[cur ^ 1][0]);

        // ---- A fragments (f32 -> bf16)
        const float* ap0 = arow0 + ks * 32;
        const float* ap1 = arow1 + ks * 32;
        if (ks < 31) { __builtin_prefetch(ap0 + 32, 0, 1); __builtin_prefetch(ap1 + 32, 0, 1); }
        v16bf a0, a1;
        {
            float4 f0 = *(const float4*)(ap0 + 0);
            float4 f1 = *(const float4*)(ap0 + 4);
            float4 f2 = *(const float4*)(ap0 + 16);
            float4 f3 = *(const float4*)(ap0 + 20);
            a0[0]=(__bf16)f0.x; a0[1]=(__bf16)f0.y; a0[2]=(__bf16)f0.z; a0[3]=(__bf16)f0.w;
            a0[4]=(__bf16)f1.x; a0[5]=(__bf16)f1.y; a0[6]=(__bf16)f1.z; a0[7]=(__bf16)f1.w;
            a0[8]=(__bf16)f2.x; a0[9]=(__bf16)f2.y; a0[10]=(__bf16)f2.z; a0[11]=(__bf16)f2.w;
            a0[12]=(__bf16)f3.x; a0[13]=(__bf16)f3.y; a0[14]=(__bf16)f3.z; a0[15]=(__bf16)f3.w;
        }
        {
            float4 f0 = *(const float4*)(ap1 + 0);
            float4 f1 = *(const float4*)(ap1 + 4);
            float4 f2 = *(const float4*)(ap1 + 16);
            float4 f3 = *(const float4*)(ap1 + 20);
            a1[0]=(__bf16)f0.x; a1[1]=(__bf16)f0.y; a1[2]=(__bf16)f0.z; a1[3]=(__bf16)f0.w;
            a1[4]=(__bf16)f1.x; a1[5]=(__bf16)f1.y; a1[6]=(__bf16)f1.z; a1[7]=(__bf16)f1.w;
            a1[8]=(__bf16)f2.x; a1[9]=(__bf16)f2.y; a1[10]=(__bf16)f2.z; a1[11]=(__bf16)f2.w;
            a1[12]=(__bf16)f3.x; a1[13]=(__bf16)f3.y; a1[14]=(__bf16)f3.z; a1[15]=(__bf16)f3.w;
        }

        // ---- gather all 4 B subtiles first (one dscnt wait), then 8 WMMAs back-to-back
        v16bf bfv[4];
#pragma unroll
        for (int s = 0; s < 4; ++s) {
            const v16bf* bp = (const v16bf*)(&smem[cur][0] +
                               ((size_t)(wave * 4 + s)) * 1024 + (size_t)lane * 32);
            bfv[s] = *bp;
        }
#pragma unroll
        for (int s = 0; s < 4; ++s) {
            c[0][s] = __builtin_amdgcn_wmma_f32_16x16x32_bf16(
                false, a0, false, bfv[s], (short)0, c[0][s], false, false);
            c[1][s] = __builtin_amdgcn_wmma_f32_16x16x32_bf16(
                false, a1, false, bfv[s], (short)0, c[1][s], false, false);
        }
    }

    // ---- epilogue: energy = tanh(c + dec_p); contrib = energy * v[n]; reduce over n.
#pragma unroll
    for (int mt = 0; mt < 2; ++mt) {
        float racc[8];
#pragma unroll
        for (int r = 0; r < 8; ++r) racc[r] = 0.0f;
#pragma unroll
        for (int s = 0; s < 4; ++s) {
            int n     = hb * 512 + (wave * 4 + s) * 16 + nlane;
            float dpv = decp[(size_t)b * Hsz + n];
            float vv  = v[n];
#pragma unroll
            for (int r = 0; r < 8; ++r) {
                float e = tanhf(c[mt][s][r] + dpv);
                racc[r] += e * vv;
            }
        }
        // C layout: VGPR r holds M=r (lanes 0-15) / M=8+r (lanes 16-31), N=lane&15.
#pragma unroll
        for (int r = 0; r < 8; ++r) {
            float x = racc[r];
            x += __shfl_xor(x, 1, 32);
            x += __shfl_xor(x, 2, 32);
            x += __shfl_xor(x, 4, 32);
            x += __shfl_xor(x, 8, 32);
            if (nlane == 0) {
                int m = trow + mt * 16 + half * 8 + r;
                atomicAdd(&scores[(size_t)b * Tsz + m], x);
            }
        }
    }
}

// ---------------------------------------------------------------- softmax
__global__ void softmax_mask(const float* __restrict__ scores, const int* __restrict__ mask,
                             float* __restrict__ attn) {
    int b   = blockIdx.x;
    int tid = threadIdx.x;            // 256 threads, 8 elements each
    __shared__ float sd[256];
    float vals[8];
    float lmax = -3.4e38f;
#pragma unroll
    for (int j = 0; j < 8; ++j) {
        int t   = tid + j * 256;
        float s = scores[(size_t)b * Tsz + t];
        if (mask[(size_t)b * Tsz + t] == 0) s = NEGVAL;
        vals[j] = s;
        lmax = fmaxf(lmax, s);
    }
    sd[tid] = lmax; __syncthreads();
    for (int st = 128; st > 0; st >>= 1) {
        if (tid < st) sd[tid] = fmaxf(sd[tid], sd[tid + st]);
        __syncthreads();
    }
    float gmax = sd[0]; __syncthreads();

    float lsum = 0.0f;
#pragma unroll
    for (int j = 0; j < 8; ++j) {
        float e = __expf(vals[j] - gmax);
        vals[j] = e;
        lsum += e;
    }
    sd[tid] = lsum; __syncthreads();
    for (int st = 128; st > 0; st >>= 1) {
        if (tid < st) sd[tid] += sd[tid + st];
        __syncthreads();
    }
    float inv = 1.0f / sd[0];
#pragma unroll
    for (int j = 0; j < 8; ++j)
        attn[(size_t)b * Tsz + tid + j * 256] = vals[j] * inv;
}

// ---------------------------------------------------------------- context
__global__ void context_kernel(const float* __restrict__ enc, const float* __restrict__ attn,
                               float* __restrict__ ctx) {
    int b  = blockIdx.x;
    int h  = blockIdx.y * 256 + threadIdx.x;
    int t0 = blockIdx.z * 256;
    float acc = 0.0f;
    const float* base = enc + ((size_t)b * Tsz + t0) * Hsz + h;
    const float* aw   = attn + (size_t)b * Tsz + t0;
    for (int t = 0; t < 256; ++t)
        acc += aw[t] * base[(size_t)t * Hsz];
    atomicAdd(&ctx[(size_t)b * Hsz + h], acc);
}

// ---------------------------------------------------------------- launcher
extern "C" void kernel_launch(void* const* d_in, const int* in_sizes, int n_in,
                              void* d_out, int out_size, void* d_ws, size_t ws_size,
                              hipStream_t stream) {
    const float* dec   = (const float*)d_in[0];   // [B,H]
    const float* enc   = (const float*)d_in[1];   // [B,T,H]
    const int*   mask  = (const int*)  d_in[2];   // [B,T]
    const float* W_enc = (const float*)d_in[3];   // [H,H]
    const float* W_dec = (const float*)d_in[4];   // [H,H]
    const float* v     = (const float*)d_in[5];   // [H]

    float* ctx_out  = (float*)d_out;                  // [B,H]
    float* attn_out = (float*)d_out + Bsz * Hsz;      // [B,T]

    // workspace carve-up
    char* ws = (char*)d_ws;
    __bf16* packedW = (__bf16*)ws;                               // 2048*512 bf16 = 2 MB
    float*  decp    = (float*)(ws + (size_t)2048 * 512 * 2);     // B*H f32 = 128 KB
    float*  scores  = decp + (size_t)Bsz * Hsz;                  // B*T f32 = 256 KB

    zero_f32<<<(Bsz * Tsz + 255) / 256, 256, 0, stream>>>(scores, Bsz * Tsz);
    zero_f32<<<(Bsz * Hsz + 255) / 256, 256, 0, stream>>>(ctx_out, Bsz * Hsz);

    pack_wenc<<<256, 256, 0, stream>>>(W_enc, packedW);
    dec_proj<<<Hsz, 256, 0, stream>>>(dec, W_dec, decp);

    // fused GEMM + tanh + v-dot -> scores ; 2048 row-blocks x 2 h-halves
    attn_scores_wmma<<<dim3(Bsz * (Tsz / 32), 2), 256, 0, stream>>>(
        enc, packedW, decp, v, scores);

    softmax_mask<<<Bsz, 256, 0, stream>>>(scores, mask, attn_out);
    context_kernel<<<dim3(Bsz, Hsz / 256, Tsz / 256), 256, 0, stream>>>(enc, attn_out, ctx_out);
}
